// HViT_UNet_59777354825966
// MI455X (gfx1250) — matrix operand order repaired and checked by reading, not compiled, not run on hardware
//
#include <hip/hip_runtime.h>
#include <hip/hip_bf16.h>
#include <math.h>

typedef _Float16 half_t;
typedef __attribute__((ext_vector_type(16))) _Float16 v16h;
typedef __attribute__((ext_vector_type(8)))  float    v8f;

#define BM 128
#define BN 128
#define BKK 32
#define LDA_S 40   // padded LDS row stride (elements) for A tile [row][k]
#define LDB_S 40   // padded LDS row stride (elements) for B tile [col][k]

// flags bits
#define F_STORE_F32 1
#define F_STORE_F16 2
#define F_GELU      4
#define F_TSTORE    8   // store C transposed: C[col*ldc + row]

union FragH { v16h v; uint4 q[2]; };
union FragF { v8f  v; float f[8]; };

__device__ __forceinline__ float gelu_exact(float x) {
    return 0.5f * x * (1.0f + erff(x * 0.70710678118654752440f));
}

// Low 32 bits of a generic pointer into LDS == LDS byte offset (shared aperture).
__device__ __forceinline__ unsigned lds_off32(const void* p) {
    return (unsigned)(unsigned long long)p;
}

// CDNA5 async copy global->LDS, 2x16B per call (offset applies to both sides).
__device__ __forceinline__ void async_copy_32B(unsigned lds, const half_t* g) {
    unsigned long long ga = (unsigned long long)g;
    asm volatile("global_load_async_to_lds_b128 %0, %1, off"
                 :: "v"(lds), "v"(ga) : "memory");
    asm volatile("global_load_async_to_lds_b128 %0, %1, off offset:16"
                 :: "v"(lds), "v"(ga) : "memory");
}

// ---------------------------------------------------------------------------
// Generic batched GEMM: C[z] = alpha * A[z] x B[z]^T-layout + bias.
// A: [M x K] row-major (lda). B: [N x K] row-major (ldb). f16 in, f32 acc.
// 128x128 block tile, 8 waves, each wave 64x32 via 4x2 WMMA 16x16x32 f16.
// Double-buffered LDS filled with GLOBAL_LOAD_ASYNC_TO_LDS_B128 (ASYNCcnt).
// ---------------------------------------------------------------------------
__global__ void __launch_bounds__(256)
gemm_f16_wmma(const half_t* __restrict__ A, long lda, int zdivA, long sA1, long sA2,
              const half_t* __restrict__ B, long ldb, int zdivB, long sB1, long sB2,
              const float* __restrict__ bias,
              float*  __restrict__ Cf, long ldcf,
              half_t* __restrict__ Ch, long ldch,
              int zdivC, long sC1, long sC2,
              int M, int N, int K, float alpha, int flags)
{
    __shared__ __align__(16) half_t As[2][BM * LDA_S];
    __shared__ __align__(16) half_t Bs[2][BN * LDB_S];

    const int  tid = threadIdx.x;
    const int  z   = blockIdx.z;
    const long m0  = (long)blockIdx.y * BM;
    const long n0  = (long)blockIdx.x * BN;

    const half_t* Ag = A + (size_t)(z / zdivA) * sA1 + (size_t)(z % zdivA) * sA2;
    const half_t* Bg = B + (size_t)(z / zdivB) * sB1 + (size_t)(z % zdivB) * sB2;
    const size_t  cOff = (size_t)(z / zdivC) * sC1 + (size_t)(z % zdivC) * sC2;

    const int wave  = tid >> 5;
    const int waveM = wave >> 2;   // 0..1 -> 64-row half
    const int waveN = wave & 3;    // 0..3 -> 32-col quarter
    const int lane  = tid & 31;
    const int lr    = lane & 15;
    const int lh    = lane >> 4;

    FragF acc[4][2];
#pragma unroll
    for (int i = 0; i < 4; ++i)
#pragma unroll
        for (int j = 0; j < 2; ++j)
            acc[i][j].v = (v8f){0.f,0.f,0.f,0.f,0.f,0.f,0.f,0.f};

    // staging coordinates: 256 threads, each moves 32B of A and 32B of B
    const int srow = tid >> 1;          // 0..127 (A row / B col)
    const int sk   = (tid & 1) * 16;    // 0 or 16
    const half_t* gA = Ag + (size_t)(m0 + srow) * lda + sk;
    const half_t* gB = Bg + (size_t)(n0 + srow) * ldb + sk;
    const unsigned lA[2] = { lds_off32(&As[0][srow * LDA_S + sk]),
                             lds_off32(&As[1][srow * LDA_S + sk]) };
    const unsigned lB[2] = { lds_off32(&Bs[0][srow * LDB_S + sk]),
                             lds_off32(&Bs[1][srow * LDB_S + sk]) };

    const int nk = K / BKK;
    // prologue: fill buffer 0 (4 async instructions per wave)
    async_copy_32B(lA[0], gA);
    async_copy_32B(lB[0], gB);

    for (int kt = 0; kt < nk; ++kt) {
        const int cur = kt & 1;
        if (kt + 1 < nk) {
            const long k1 = (long)(kt + 1) * BKK;
            async_copy_32B(lA[cur ^ 1], gA + k1);
            async_copy_32B(lB[cur ^ 1], gB + k1);
            // 4 newer ops outstanding; <=4 left pending => current tile landed
            asm volatile("s_wait_asynccnt 0x4" ::: "memory");
        } else {
            asm volatile("s_wait_asynccnt 0x0" ::: "memory");
        }
        __syncthreads();   // current buffer visible to all waves

        const half_t* As_c = As[cur];
        const half_t* Bs_c = Bs[cur];

        FragH fa[4], fb[2];
#pragma unroll
        for (int mi = 0; mi < 4; ++mi) {
            const half_t* ap = &As_c[(waveM * 64 + mi * 16 + lr) * LDA_S];
            fa[mi].q[0] = *(const uint4*)(ap + lh * 8);        // K = lh*8 .. +7
            fa[mi].q[1] = *(const uint4*)(ap + 16 + lh * 8);   // K = 16+lh*8 .. +7
        }
#pragma unroll
        for (int ni = 0; ni < 2; ++ni) {
            const half_t* bp = &Bs_c[(waveN * 32 + ni * 16 + lr) * LDB_S];
            fb[ni].q[0] = *(const uint4*)(bp + lh * 8);
            fb[ni].q[1] = *(const uint4*)(bp + 16 + lh * 8);
        }
#pragma unroll
        for (int mi = 0; mi < 4; ++mi)
#pragma unroll
            for (int ni = 0; ni < 2; ++ni)
                acc[mi][ni].v = __builtin_amdgcn_wmma_f32_16x16x32_f16(
                    false, fa[mi].v, false, fb[ni].v,
                    (short)0, acc[mi][ni].v, false, false);
        __syncthreads();   // all waves done reading 'cur' before it is refilled
    }

    // ---- epilogue ----
#pragma unroll
    for (int mi = 0; mi < 4; ++mi) {
#pragma unroll
        for (int ni = 0; ni < 2; ++ni) {
            const long col = n0 + waveN * 32 + ni * 16 + lr;
            const float bv = bias ? bias[col] : 0.0f;
#pragma unroll
            for (int j = 0; j < 8; ++j) {
                const long row = m0 + waveM * 64 + mi * 16 + lh * 8 + j;
                float c = acc[mi][ni].f[j] * alpha + bv;
                if (flags & F_GELU) c = gelu_exact(c);
                const size_t oF = (flags & F_TSTORE) ? (size_t)col * ldcf + row
                                                     : (size_t)row * ldcf + col;
                const size_t oH = (flags & F_TSTORE) ? (size_t)col * ldch + row
                                                     : (size_t)row * ldch + col;
                if (flags & F_STORE_F32) Cf[cOff + oF] = c;
                if (flags & F_STORE_F16) Ch[cOff + oH] = (half_t)c;
            }
        }
    }
}

// ---------------------------------------------------------------------------
// helpers
// ---------------------------------------------------------------------------
// src [L][R][C] f32 -> dst [L][C][R] f16 (transpose + convert), coalesced writes
__global__ void __launch_bounds__(256)
transpose_f32_to_f16_kern(const float* __restrict__ src, half_t* __restrict__ dst,
                          int R, int C, long total)
{
    long idx = (long)blockIdx.x * 256 + threadIdx.x;
    if (idx >= total) return;
    long rc = (long)R * C;
    int  l   = (int)(idx / rc);
    long rem = idx - (long)l * rc;
    int  n   = (int)(rem / R);
    int  k   = (int)(rem - (long)n * R);
    dst[idx] = (half_t)src[(size_t)l * rc + (size_t)k * C + n];
}

// enc0[b,n,d] = X[b, (n/16)*16 + d/16, (n%16)*16 + d%16, 0] + pos[n,d]  (f16 out)
__global__ void __launch_bounds__(256)
patch_encode_kern(const float* __restrict__ X, const float* __restrict__ pos,
                  half_t* __restrict__ peH)
{
    long idx = (long)blockIdx.x * 256 + threadIdx.x;   // 8192*256
    int b = (int)(idx >> 16);
    int rem = (int)(idx & 65535);
    int n = rem >> 8, d = rem & 255;
    int y = ((n >> 4) << 4) + (d >> 4);
    int x = ((n & 15) << 4) + (d & 15);
    float v = X[((size_t)b << 16) + y * 256 + x] + pos[rem];
    peH[idx] = (half_t)v;
}

// one wave32 per 256-wide row
__global__ void __launch_bounds__(256)
softmax_rows_kern(const float* __restrict__ S, half_t* __restrict__ P)
{
    const int wave = threadIdx.x >> 5;
    const int lane = threadIdx.x & 31;
    const size_t row = (size_t)blockIdx.x * 8 + wave;
    const float* s = S + row * 256;
    half_t* p = P + row * 256;
    float v[8];
    float m = -3.4e38f;
#pragma unroll
    for (int j = 0; j < 8; ++j) { v[j] = s[j * 32 + lane]; m = fmaxf(m, v[j]); }
#pragma unroll
    for (int o = 16; o > 0; o >>= 1) m = fmaxf(m, __shfl_xor(m, o, 32));
    float sum = 0.f;
#pragma unroll
    for (int j = 0; j < 8; ++j) { v[j] = __expf(v[j] - m); sum += v[j]; }
#pragma unroll
    for (int o = 16; o > 0; o >>= 1) sum += __shfl_xor(sum, o, 32);
    const float inv = 1.0f / sum;
#pragma unroll
    for (int j = 0; j < 8; ++j) p[j * 32 + lane] = (half_t)(v[j] * inv);
}

// enc = LN(enc + a) * g + b  (per 256-wide row), also emit f16 copy
__global__ void __launch_bounds__(256)
residual_ln_kern(float* __restrict__ enc, const float* __restrict__ a,
                 const float* __restrict__ g, const float* __restrict__ b,
                 half_t* __restrict__ encH)
{
    __shared__ float red[256];
    const size_t row = blockIdx.x;
    const int t = threadIdx.x;
    const size_t off = row * 256 + t;
    const float x = enc[off] + a[off];
    red[t] = x; __syncthreads();
    for (int s = 128; s > 0; s >>= 1) { if (t < s) red[t] += red[t + s]; __syncthreads(); }
    const float mu = red[0] * (1.0f / 256.0f);
    __syncthreads();
    const float d = x - mu;
    red[t] = d * d; __syncthreads();
    for (int s = 128; s > 0; s >>= 1) { if (t < s) red[t] += red[t + s]; __syncthreads(); }
    const float var = red[0] * (1.0f / 256.0f);
    const float y = d * rsqrtf(var + 1e-3f) * g[t] + b[t];
    enc[off] = y;
    encH[off] = (half_t)y;
}

// unpatch(P1=16) then patchify(P2=8): permutation enc[8192,256] -> out[32,1024,64]
__global__ void __launch_bounds__(256)
resample_kern(const float* __restrict__ enc, float* __restrict__ out)
{
    long idx = (long)blockIdx.x * 256 + threadIdx.x;   // 32*1024*64
    int b = (int)(idx >> 16);
    int rem = (int)(idx & 65535);
    int n2 = rem >> 6, d2 = rem & 63;
    int y = ((n2 >> 5) << 3) + (d2 >> 3);
    int x = ((n2 & 31) << 3) + (d2 & 7);
    int n1 = ((y >> 4) << 4) + (x >> 4);
    int d1 = ((y & 15) << 4) + (x & 15);
    out[idx] = enc[((size_t)b << 16) + n1 * 256 + d1];
}

// ---------------------------------------------------------------------------
extern "C" void kernel_launch(void* const* d_in, const int* in_sizes, int n_in,
                              void* d_out, int out_size, void* d_ws, size_t ws_size,
                              hipStream_t stream)
{
    const float* X    = (const float*)d_in[0];
    const float* pos  = (const float*)d_in[1];
    const float* W_in = (const float*)d_in[2];
    const float* b_in = (const float*)d_in[3];
    const float* Wq   = (const float*)d_in[4];
    const float* bq   = (const float*)d_in[5];
    const float* Wk   = (const float*)d_in[6];
    const float* bk   = (const float*)d_in[7];
    const float* Wv   = (const float*)d_in[8];
    const float* bv   = (const float*)d_in[9];
    const float* Wo   = (const float*)d_in[10];
    const float* bo   = (const float*)d_in[11];
    const float* ln1g = (const float*)d_in[12];
    const float* ln1b = (const float*)d_in[13];
    const float* ln2g = (const float*)d_in[14];
    const float* ln2b = (const float*)d_in[15];
    const float* W1   = (const float*)d_in[16];
    const float* b1   = (const float*)d_in[17];
    const float* W2   = (const float*)d_in[18];
    const float* b2   = (const float*)d_in[19];

    const long BN1 = 8192;   // B * N1
    const long D1  = 256, NHKD = 2048, HID = 1024;
    const long Z   = 256;    // B * NH

    char* ws = (char*)d_ws;
    size_t off = 0;
    auto alloc = [&](size_t bytes) -> void* {
        void* p = ws + off;
        off = (off + bytes + 255) & ~(size_t)255;
        return p;
    };
    float*  encF = (float*)  alloc(BN1 * D1 * 4);
    float*  aF   = (float*)  alloc(BN1 * D1 * 4);
    half_t* encH = (half_t*) alloc(BN1 * D1 * 2);
    half_t* peH  = (half_t*) alloc(BN1 * D1 * 2);
    half_t* qH   = (half_t*) alloc(BN1 * NHKD * 2);   // row-major [bn][hk]
    half_t* kH   = (half_t*) alloc(BN1 * NHKD * 2);   // row-major [bn][hk]
    half_t* vtH  = (half_t*) alloc(BN1 * NHKD * 2);   // transposed [b][hk][n]
    float*  S    = (float*)  alloc(Z * 256 * 256 * 4);
    half_t* P    = (half_t*) alloc(Z * 256 * 256 * 2);
    half_t* WinT = (half_t*) alloc(256 * 256 * 2);              // [256 n][256 k]
    half_t* WqT  = (half_t*) alloc(8L * 2048 * 256 * 2);        // [l][2048 n][256 k]
    half_t* WkT  = (half_t*) alloc(8L * 2048 * 256 * 2);
    half_t* WvT  = (half_t*) alloc(8L * 2048 * 256 * 2);
    half_t* WoT  = (half_t*) alloc(8L * 256 * 2048 * 2);        // [l][256 n][2048 k]
    half_t* W1T  = (half_t*) alloc(8L * 1024 * 256 * 2);        // [l][1024 n][256 k]
    half_t* W2T  = (half_t*) alloc(8L * 256 * 1024 * 2);        // [l][256 n][1024 k]
    half_t* oH = qH;            // O reuses Q (Q dead once logits exist)
    half_t* fH = (half_t*)S;    // FFN hidden reuses logits buffer

    auto tconv = [&](const float* s, half_t* d, int R, int C, int L) {
        long n = (long)L * R * C;
        transpose_f32_to_f16_kern<<<dim3((unsigned)((n + 255) / 256)), dim3(256), 0, stream>>>(
            s, d, R, C, n);
    };
    tconv(W_in, WinT, 256, 256, 1);
    tconv(Wq, WqT, 256, 2048, 8);
    tconv(Wk, WkT, 256, 2048, 8);
    tconv(Wv, WvT, 256, 2048, 8);
    tconv(Wo, WoT, 2048, 256, 8);
    tconv(W1, W1T, 256, 1024, 8);
    tconv(W2, W2T, 1024, 256, 8);

    patch_encode_kern<<<dim3(8192), dim3(256), 0, stream>>>(X, pos, peH);

    auto gemm = [&](const half_t* A, long lda, int zdA, long sA1, long sA2,
                    const half_t* B, long ldb, int zdB, long sB1, long sB2,
                    const float* bias, float* Cf, long ldcf, half_t* Ch, long ldch,
                    int zdC, long sC1, long sC2,
                    int M, int N, int K, int batch, float alpha, int flags) {
        dim3 grid((unsigned)(N / BN), (unsigned)(M / BM), (unsigned)batch);
        gemm_f16_wmma<<<grid, dim3(256), 0, stream>>>(
            A, lda, zdA, sA1, sA2, B, ldb, zdB, sB1, sB2, bias,
            Cf, ldcf, Ch, ldch, zdC, sC1, sC2, M, N, K, alpha, flags);
    };

    // enc = pe @ W_in + b_in  -> f32 + f16
    gemm(peH, D1, 1, 0, 0, WinT, D1, 1, 0, 0, b_in,
         encF, D1, encH, D1, 1, 0, 0,
         (int)BN1, (int)D1, (int)D1, 1, 1.0f, F_STORE_F32 | F_STORE_F16);

    for (int i = 0; i < 8; ++i) {
        const half_t* WqT_i = WqT + (size_t)i * 2048 * 256;
        const half_t* WkT_i = WkT + (size_t)i * 2048 * 256;
        const half_t* WvT_i = WvT + (size_t)i * 2048 * 256;
        const half_t* WoT_i = WoT + (size_t)i * 256 * 2048;
        const half_t* W1T_i = W1T + (size_t)i * 1024 * 256;
        const half_t* W2T_i = W2T + (size_t)i * 256 * 1024;
        const float* bq_i = bq + (size_t)i * 2048;
        const float* bk_i = bk + (size_t)i * 2048;
        const float* bv_i = bv + (size_t)i * 2048;
        const float* bo_i = bo + (size_t)i * 256;
        const float* b1_i = b1 + (size_t)i * 1024;
        const float* b2_i = b2 + (size_t)i * 256;

        // Q/K projections: [8192,256] x [2048,256]^T -> f16 row-major
        gemm(encH, D1, 1, 0, 0, WqT_i, D1, 1, 0, 0, bq_i,
             nullptr, 0, qH, NHKD, 1, 0, 0, (int)BN1, (int)NHKD, (int)D1, 1, 1.0f,
             F_STORE_F16);
        gemm(encH, D1, 1, 0, 0, WkT_i, D1, 1, 0, 0, bk_i,
             nullptr, 0, kH, NHKD, 1, 0, 0, (int)BN1, (int)NHKD, (int)D1, 1, 1.0f,
             F_STORE_F16);
        // V projection, batched over b, stored transposed: vtH[b][hk][n]
        gemm(encH, D1, 1, 65536, 0, WvT_i, D1, 1, 0, 0, bv_i,
             nullptr, 0, vtH, 256, 1, 524288, 0,
             256, (int)NHKD, (int)D1, 32, 1.0f, F_STORE_F16 | F_TSTORE);

        // logits: per (b,h): S = (Q Kt) / 16  [256x256], batch Z=256
        gemm(qH, NHKD, 8, 256L * 2048, 256, kH, NHKD, 8, 256L * 2048, 256, nullptr,
             S, 256, nullptr, 0, 1, 65536, 0,
             256, 256, 256, (int)Z, 0.0625f, F_STORE_F32);

        softmax_rows_kern<<<dim3(8192), dim3(256), 0, stream>>>(S, P);

        // O = P @ V: A=P [256x256], B=Vt[hk-block][n] -> oH[b, n, h*256+kd] f16
        gemm(P, 256, 1, 65536, 0, vtH, 256, 1, 65536, 0, nullptr,
             nullptr, 0, oH, NHKD, 8, 256L * 2048, 256,
             256, 256, 256, (int)Z, 1.0f, F_STORE_F16);

        // out projection: [8192,2048] x [256,2048]^T + bo -> aF f32
        gemm(oH, NHKD, 1, 0, 0, WoT_i, NHKD, 1, 0, 0, bo_i,
             aF, D1, nullptr, 0, 1, 0, 0, (int)BN1, (int)D1, (int)NHKD, 1, 1.0f,
             F_STORE_F32);

        residual_ln_kern<<<dim3(8192), dim3(256), 0, stream>>>(
            encF, aF, ln1g + (size_t)i * 256, ln1b + (size_t)i * 256, encH);

        // FFN1: gelu(enc @ W1 + b1) -> f16
        gemm(encH, D1, 1, 0, 0, W1T_i, D1, 1, 0, 0, b1_i,
             nullptr, 0, fH, HID, 1, 0, 0, (int)BN1, (int)HID, (int)D1, 1, 1.0f,
             F_STORE_F16 | F_GELU);
        // FFN2: gelu(f @ W2 + b2) -> f32
        gemm(fH, HID, 1, 0, 0, W2T_i, HID, 1, 0, 0, b2_i,
             aF, D1, nullptr, 0, 1, 0, 0, (int)BN1, (int)D1, (int)HID, 1, 1.0f,
             F_STORE_F32 | F_GELU);

        residual_ln_kern<<<dim3(8192), dim3(256), 0, stream>>>(
            encF, aF, ln2g + (size_t)i * 256, ln2b + (size_t)i * 256, encH);
    }

    resample_kern<<<dim3(8192), dim3(256), 0, stream>>>(encF, (float*)d_out);

    (void)in_sizes; (void)n_in; (void)out_size; (void)ws_size;
}
